// Encoder_74397423501762
// MI455X (gfx1250) — matrix operand order repaired and checked by reading, not compiled, not run on hardware
//
#include <hip/hip_runtime.h>
#include <cstdint>

// ---------------------------------------------------------------------------
// CDNA5 / gfx1250 transformer encoder block.
// bf16 WMMA (v_wmma_f32_16x16x32_bf16) for all matmuls, f32 accum/softmax/LN.
// Tensor Data Mover (tensor_load_to_lds) stages A-panels / Q / K tiles into
// LDS with descriptor-driven padding that reproduces the bank-conflict-free
// strides; completion via s_wait_tensorcnt.
// ---------------------------------------------------------------------------

typedef __bf16 bf16;
typedef bf16  v16bf __attribute__((ext_vector_type(16)));
typedef float v8f   __attribute__((ext_vector_type(8)));
typedef unsigned int u32x4 __attribute__((ext_vector_type(4)));
typedef int i32x8 __attribute__((ext_vector_type(8)));
typedef int i32x4 __attribute__((ext_vector_type(4)));

union FragB16 { v16bf v; unsigned u[8]; };

__device__ __forceinline__ v8f wmma_bf16(v16bf a, v16bf b, v8f c) {
  return __builtin_amdgcn_wmma_f32_16x16x32_bf16(false, a, false, b, (short)0, c,
                                                 false, false);
}

// A-matrix 16x32 bf16 fragment from row-major LDS tile:
// lanes 0-15: VGPR0-7 = K 0..15 (2/VGPR)
// lanes 16-31: VGPR0-7 = K 16..31
__device__ __forceinline__ v16bf load_a_frag(const bf16* tile, int stride, int m, int hl) {
  FragB16 f;
  const bf16* row = tile + m * stride;
#pragma unroll
  for (int v = 0; v < 8; ++v) {
    int k0 = hl * 16 + 2 * v;
    f.u[v]     = *(const unsigned*)(row + k0);
  }
  return f.v;
}

// B-matrix 32x16 bf16 fragment from *transposed* LDS tile Bt[n][k]:
// lane n owns column n; VGPR v holds K = hl*16 + 2v, +1.
__device__ __forceinline__ v16bf load_b_frag(const bf16* tileT, int stride, int n, int hl) {
  FragB16 f;
  const bf16* col = tileT + n * stride + hl * 16;
#pragma unroll
  for (int v = 0; v < 8; ++v) f.u[v] = *(const unsigned*)(col + 2 * v);
  return f.v;
}

// ---------------------------------------------------------------------------
// Tensor Data Mover: DMA a 2D tile (tile_d0_dw DWORDs wide x tile_d1 rows) of
// a row-major tensor (row stride stride_dw DWORDs) into LDS at lds_off, with
// LDS row padding (pad codes per ISA: interval 3->16DW, 4->32DW; amount
// 3 -> 4 DWORDs).  D# layout per CDNA5 ISA.
__device__ __forceinline__ void tdm_load_2d(unsigned lds_off, const void* gp,
                                            unsigned tile_d0_dw, unsigned tile_d1,
                                            unsigned stride_dw,
                                            unsigned pad_interval, unsigned pad_amount) {
  unsigned long long ga = (unsigned long long)(uintptr_t)gp;
  const unsigned tensor_d1 = 1u << 20;               // generous; tiles in-bounds
  u32x4 g0;
  g0[0] = 1u;                                        // count=1, user descriptor
  g0[1] = lds_off;                                   // lds_addr (bytes)
  g0[2] = (unsigned)(ga & 0xffffffffu);              // global_addr[31:0]
  g0[3] = (unsigned)((ga >> 32) & 0x01ffffffu)       // global_addr[56:32]
          | (2u << 30);                              // type=2 ("image")
  i32x8 g1;
  g1[0] = (int)((2u << 16) |                         // data_size = 4B units
                (1u << 20) |                         // pad_enable
                (pad_interval << 22) | (pad_amount << 25));
  g1[1] = (int)(tile_d0_dw << 16);                   // tensor_dim0[15:0] @ 63:48
  g1[2] = (int)((tile_d0_dw >> 16) |                 // tensor_dim0[31:16] @ 79:64
                ((tensor_d1 & 0xffffu) << 16));      // tensor_dim1[15:0] @ 95:80
  g1[3] = (int)((tensor_d1 >> 16) |                  // tensor_dim1[31:16] @ 111:96
                (tile_d0_dw << 16));                 // tile_dim0 @ 127:112
  g1[4] = (int)(tile_d1 & 0xffffu);                  // tile_dim1 (tile_dim2 = 0)
  g1[5] = (int)stride_dw;                            // tensor_dim0_stride[31:0]
  g1[6] = 0;
  g1[7] = 0;
  i32x4 z4 = {0, 0, 0, 0};
  i32x8 z8 = {0, 0, 0, 0, 0, 0, 0, 0};
  __builtin_amdgcn_tensor_load_to_lds(g0, g1, z4, z4, z8, 0);
}

// ---------------------------------------------------------------------------
__global__ __launch_bounds__(256) void cvt_bf16(const float* __restrict__ in,
                                                bf16* __restrict__ out, int n) {
  for (int i = blockIdx.x * 256 + threadIdx.x; i < n; i += gridDim.x * 256)
    out[i] = (bf16)in[i];
}

// ---------------------------------------------------------------------------
// kqv = x(4096x512) @ Wkqv(512x1536); split into k/q/v buffers [N,H,L,64] bf16.
// Scale 1/sqrt(64) folded into q.
__global__ __launch_bounds__(256) void gemm_kqv(const bf16* __restrict__ A,
                                                const bf16* __restrict__ B,
                                                bf16* __restrict__ kq,
                                                bf16* __restrict__ qq,
                                                bf16* __restrict__ vq) {
  __shared__ bf16 sA[128 * 40];
  __shared__ bf16 sBt[64 * 40];
  int tid = threadIdx.x, wv = tid >> 5, lane = tid & 31, hl = lane >> 4, n = lane & 15;
  int mb = blockIdx.x * 128, nb = blockIdx.y * 64;
  v8f acc[4] = {};
  for (int k0 = 0; k0 < 512; k0 += 32) {
    { // 32x64 B panel, transposed into sBt (all threads)
      int r = tid >> 3, c0 = (tid & 7) * 8;
      uint4 d = *(const uint4*)(B + (size_t)(k0 + r) * 1536 + nb + c0);
      const bf16* dh = (const bf16*)&d;
#pragma unroll
      for (int i = 0; i < 8; ++i) sBt[(c0 + i) * 40 + r] = dh[i];
    }
    if (wv == 0) { // 128x32 A panel via TDM: 64B rows + 16B pad -> stride 40
      tdm_load_2d((unsigned)(uintptr_t)sA, A + (size_t)mb * 512 + k0,
                  /*tile_d0_dw=*/16, /*tile_d1=*/128, /*stride_dw=*/256,
                  /*pad_interval=*/3, /*pad_amount=*/3);
      __builtin_amdgcn_s_wait_tensorcnt(0);
    }
    __syncthreads();
    v16bf a = load_a_frag(sA + wv * 16 * 40, 40, n, hl);
#pragma unroll
    for (int f = 0; f < 4; ++f) {
      v16bf b = load_b_frag(sBt + f * 16 * 40, 40, n, hl);
      acc[f] = wmma_bf16(a, b, acc[f]);
    }
    __syncthreads();
  }
#pragma unroll
  for (int f = 0; f < 4; ++f) {
    int colb = nb + f * 16;
    int seg = colb >> 9;            // 0=k 1=q 2=v
    int head = (colb >> 6) & 7;
    int hd = (colb & 63) + n;
    bf16* dst = (seg == 0) ? kq : (seg == 1) ? qq : vq;
    float scl = (seg == 1) ? 0.125f : 1.0f;
#pragma unroll
    for (int r = 0; r < 8; ++r) {
      int gr = mb + wv * 16 + hl * 8 + r;
      int batch = gr >> 11, l = gr & 2047;
      dst[(((size_t)batch * 8 + head) * 2048 + l) * 64 + hd] = (bf16)(acc[f][r] * scl);
    }
  }
}

// ---------------------------------------------------------------------------
// Flash attention per (batch,head). Block: 128 query rows, streams 32-key tiles.
__global__ __launch_bounds__(256) void flash_attn(const bf16* __restrict__ Q,
                                                  const bf16* __restrict__ K,
                                                  const bf16* __restrict__ V,
                                                  const float* __restrict__ mask,
                                                  bf16* __restrict__ O) {
  __shared__ bf16 sQ[128 * 72];
  __shared__ bf16 sK[32 * 72];
  __shared__ bf16 sVt[64 * 40];
  __shared__ bf16 sP[8 * 16 * 40];
  int tid = threadIdx.x, wv = tid >> 5, lane = tid & 31, hl = lane >> 4, n = lane & 15;
  int gy = blockIdx.y;              // batch*8 + head
  int q0 = blockIdx.x * 128;
  const bf16* qg = Q + ((size_t)gy * 2048 + q0) * 64;
  const bf16* kg = K + (size_t)gy * 2048 * 64;
  const bf16* vg = V + (size_t)gy * 2048 * 64;
  const float* mg = mask + ((size_t)gy * 2048 + q0) * 2048;

  if (wv == 0) {                    // stage 128x64 Q via TDM: 128B rows + 16B pad
    tdm_load_2d((unsigned)(uintptr_t)sQ, qg,
                /*tile_d0_dw=*/32, /*tile_d1=*/128, /*stride_dw=*/32,
                /*pad_interval=*/4, /*pad_amount=*/3);
    __builtin_amdgcn_s_wait_tensorcnt(0);
  }
  __syncthreads();
  v16bf qf0 = load_a_frag(sQ + wv * 16 * 72, 72, n, hl);
  v16bf qf1 = load_a_frag(sQ + wv * 16 * 72 + 32, 72, n, hl);

  v8f o[4] = {};
  float mrun[8], lrun[8];
#pragma unroll
  for (int r = 0; r < 8; ++r) { mrun[r] = -1e30f; lrun[r] = 0.f; }
  bf16* pP = sP + wv * (16 * 40);

  for (int kb = 0; kb < 2048; kb += 32) {
    { // stage V tile transposed (all threads)
      int r = tid >> 3, c = (tid & 7) * 8;
      uint4 d = *(const uint4*)(vg + (size_t)(kb + r) * 64 + c);
      const bf16* dh = (const bf16*)&d;
#pragma unroll
      for (int i = 0; i < 8; ++i) sVt[(c + i) * 40 + r] = dh[i];
    }
    if (wv == 0) {                  // stage 32x64 K tile via TDM
      tdm_load_2d((unsigned)(uintptr_t)sK, kg + (size_t)kb * 64,
                  /*tile_d0_dw=*/32, /*tile_d1=*/32, /*stride_dw=*/32,
                  /*pad_interval=*/4, /*pad_amount=*/3);
      __builtin_amdgcn_s_wait_tensorcnt(0);
    }
    __syncthreads();
    // scores: S(16x32) = Qw(16x64) x K^T ; sK is already the transposed-B layout
    v8f s0v = {}, s1v = {};
    {
      v16bf b00 = load_b_frag(sK, 72, n, hl);
      v16bf b01 = load_b_frag(sK + 32, 72, n, hl);
      v16bf b10 = load_b_frag(sK + 16 * 72, 72, n, hl);
      v16bf b11 = load_b_frag(sK + 16 * 72 + 32, 72, n, hl);
      s0v = wmma_bf16(qf0, b00, s0v); s0v = wmma_bf16(qf1, b01, s0v);
      s1v = wmma_bf16(qf0, b10, s1v); s1v = wmma_bf16(qf1, b11, s1v);
    }
    // online softmax over this 32-key block
#pragma unroll
    for (int r = 0; r < 8; ++r) {
      const float* mrow = mg + (size_t)(wv * 16 + hl * 8 + r) * 2048 + kb;
      float s0 = s0v[r] + mrow[n];
      float s1 = s1v[r] + mrow[16 + n];
      float mx = fmaxf(s0, s1);
#pragma unroll
      for (int d = 1; d < 16; d <<= 1) mx = fmaxf(mx, __shfl_xor(mx, d, 16));
      float nm = fmaxf(mrun[r], mx);
      float corr = __expf(mrun[r] - nm);
      float p0 = __expf(s0 - nm), p1 = __expf(s1 - nm);
      float rs = p0 + p1;
#pragma unroll
      for (int d = 1; d < 16; d <<= 1) rs += __shfl_xor(rs, d, 16);
      lrun[r] = lrun[r] * corr + rs;
      mrun[r] = nm;
#pragma unroll
      for (int f = 0; f < 4; ++f) o[f][r] *= corr;
      pP[(hl * 8 + r) * 40 + n]      = (bf16)p0;   // re-layout P via LDS
      pP[(hl * 8 + r) * 40 + 16 + n] = (bf16)p1;
    }
    __syncthreads();
    v16bf pa = load_a_frag(pP, 40, n, hl);
#pragma unroll
    for (int f = 0; f < 4; ++f) {
      v16bf b = load_b_frag(sVt + f * 16 * 40, 40, n, hl);
      o[f] = wmma_bf16(pa, b, o[f]);
    }
    __syncthreads();
  }
  int batch = gy >> 3, head = gy & 7;
#pragma unroll
  for (int r = 0; r < 8; ++r) {
    float inv = 1.0f / lrun[r];
    int l = q0 + wv * 16 + hl * 8 + r;
#pragma unroll
    for (int f = 0; f < 4; ++f)
      O[((size_t)batch * 2048 + l) * 512 + head * 64 + f * 16 + n] = (bf16)(o[f][r] * inv);
  }
}

// ---------------------------------------------------------------------------
// H = gelu(attn(4096x512) @ W1(512x2048) + b1), exact GELU, bf16 out.
__global__ __launch_bounds__(256) void gemm_ffn1(const bf16* __restrict__ A,
                                                 const bf16* __restrict__ B,
                                                 const float* __restrict__ b1,
                                                 bf16* __restrict__ H) {
  __shared__ bf16 sA[128 * 40];
  __shared__ bf16 sBt[64 * 40];
  int tid = threadIdx.x, wv = tid >> 5, lane = tid & 31, hl = lane >> 4, n = lane & 15;
  int mb = blockIdx.x * 128, nb = blockIdx.y * 64;
  v8f acc[4] = {};
  for (int k0 = 0; k0 < 512; k0 += 32) {
    { int r = tid >> 3, c0 = (tid & 7) * 8;
      uint4 d = *(const uint4*)(B + (size_t)(k0 + r) * 2048 + nb + c0);
      const bf16* dh = (const bf16*)&d;
#pragma unroll
      for (int i = 0; i < 8; ++i) sBt[(c0 + i) * 40 + r] = dh[i];
    }
    if (wv == 0) {
      tdm_load_2d((unsigned)(uintptr_t)sA, A + (size_t)mb * 512 + k0,
                  16, 128, 256, 3, 3);
      __builtin_amdgcn_s_wait_tensorcnt(0);
    }
    __syncthreads();
    v16bf a = load_a_frag(sA + wv * 16 * 40, 40, n, hl);
#pragma unroll
    for (int f = 0; f < 4; ++f) {
      v16bf b = load_b_frag(sBt + f * 16 * 40, 40, n, hl);
      acc[f] = wmma_bf16(a, b, acc[f]);
    }
    __syncthreads();
  }
#pragma unroll
  for (int f = 0; f < 4; ++f) {
    int col = nb + f * 16 + n;
    float bv = b1[col];
#pragma unroll
    for (int r = 0; r < 8; ++r) {
      int gr = mb + wv * 16 + hl * 8 + r;
      float xv = acc[f][r] + bv;
      float g = 0.5f * xv * (1.0f + erff(xv * 0.70710678118654752f));
      H[(size_t)gr * 2048 + col] = (bf16)g;
    }
  }
}

// ---------------------------------------------------------------------------
// ffn = H(4096x2048) @ W2(2048x256) + b2 ; LayerNorm(256) * gamma + beta ;
// + residual mean(x.reshape(...,2,256)). f32 out. Wave computes 16x256.
__global__ __launch_bounds__(256) void gemm_ffn2_ln(const bf16* __restrict__ A,
                                                    const bf16* __restrict__ B,
                                                    const float* __restrict__ b2,
                                                    const float* __restrict__ gamma,
                                                    const float* __restrict__ beta,
                                                    const float* __restrict__ x,
                                                    float* __restrict__ out) {
  __shared__ bf16 sA[128 * 40];
  __shared__ bf16 sBt[256 * 40];
  int tid = threadIdx.x, wv = tid >> 5, lane = tid & 31, hl = lane >> 4, n = lane & 15;
  int mb = blockIdx.x * 128;
  v8f acc[16] = {};
  for (int k0 = 0; k0 < 2048; k0 += 32) {
#pragma unroll
    for (int i = 0; i < 4; ++i) {   // 32x256 B panel transposed (all threads)
      int cc = tid * 4 + i; int r = cc >> 5, c0 = (cc & 31) * 8;
      uint4 d = *(const uint4*)(B + (size_t)(k0 + r) * 256 + c0);
      const bf16* dh = (const bf16*)&d;
#pragma unroll
      for (int j = 0; j < 8; ++j) sBt[(c0 + j) * 40 + r] = dh[j];
    }
    if (wv == 0) {
      tdm_load_2d((unsigned)(uintptr_t)sA, A + (size_t)mb * 2048 + k0,
                  16, 128, 1024, 3, 3);
      __builtin_amdgcn_s_wait_tensorcnt(0);
    }
    __syncthreads();
    v16bf a = load_a_frag(sA + wv * 16 * 40, 40, n, hl);
#pragma unroll
    for (int f = 0; f < 16; ++f) {
      v16bf b = load_b_frag(sBt + f * 16 * 40, 40, n, hl);
      acc[f] = wmma_bf16(a, b, acc[f]);
    }
    __syncthreads();
  }
  float s1[8] = {}, s2[8] = {};
#pragma unroll
  for (int f = 0; f < 16; ++f) {
    float bv = b2[f * 16 + n];
#pragma unroll
    for (int r = 0; r < 8; ++r) {
      float t = acc[f][r] + bv;
      acc[f][r] = t;
      s1[r] += t; s2[r] += t * t;
    }
  }
#pragma unroll
  for (int r = 0; r < 8; ++r) {
#pragma unroll
    for (int d = 1; d < 16; d <<= 1) {
      s1[r] += __shfl_xor(s1[r], d, 16);
      s2[r] += __shfl_xor(s2[r], d, 16);
    }
  }
#pragma unroll
  for (int r = 0; r < 8; ++r) {
    float mu = s1[r] * (1.0f / 256.0f);
    float var = s2[r] * (1.0f / 256.0f) - mu * mu;
    float rsq = rsqrtf(var + 1e-5f);
    int gr = mb + wv * 16 + hl * 8 + r;
#pragma unroll
    for (int f = 0; f < 16; ++f) {
      int col = f * 16 + n;
      float res = 0.5f * (x[(size_t)gr * 512 + col] + x[(size_t)gr * 512 + 256 + col]);
      out[(size_t)gr * 256 + col] =
          (acc[f][r] - mu) * rsq * gamma[col] + beta[col] + res;
    }
  }
}

// ---------------------------------------------------------------------------
extern "C" void kernel_launch(void* const* d_in, const int* in_sizes, int n_in,
                              void* d_out, int out_size, void* d_ws, size_t ws_size,
                              hipStream_t stream) {
  (void)in_sizes; (void)n_in; (void)out_size; (void)ws_size;
  const float* x     = (const float*)d_in[0];
  const float* mask  = (const float*)d_in[1];
  const float* Wkqv  = (const float*)d_in[2];
  const float* W1    = (const float*)d_in[3];
  const float* b1    = (const float*)d_in[4];
  const float* W2    = (const float*)d_in[5];
  const float* b2    = (const float*)d_in[6];
  const float* gamma = (const float*)d_in[7];
  const float* beta  = (const float*)d_in[8];
  float* out = (float*)d_out;

  char* w = (char*)d_ws;
  bf16* xb    = (bf16*)(w);                 // 4096x512
  bf16* wkqvb = (bf16*)(w + 4194304);       // 512x1536
  bf16* w1b   = (bf16*)(w + 5767168);       // 512x2048
  bf16* w2b   = (bf16*)(w + 7864320);       // 2048x256
  bf16* qb    = (bf16*)(w + 8912896);       // [2,8,2048,64]
  bf16* kb    = (bf16*)(w + 13107200);
  bf16* vb    = (bf16*)(w + 17301504);
  bf16* attn  = (bf16*)(w + 21495808);      // 4096x512
  bf16* hb    = (bf16*)(w + 25690112);      // 4096x2048

  cvt_bf16<<<1024, 256, 0, stream>>>(x, xb, 4096 * 512);
  cvt_bf16<<<1024, 256, 0, stream>>>(Wkqv, wkqvb, 512 * 1536);
  cvt_bf16<<<1024, 256, 0, stream>>>(W1, w1b, 512 * 2048);
  cvt_bf16<<<256, 256, 0, stream>>>(W2, w2b, 2048 * 256);
  gemm_kqv<<<dim3(32, 24), 256, 0, stream>>>(xb, wkqvb, kb, qb, vb);
  flash_attn<<<dim3(16, 16), 256, 0, stream>>>(qb, kb, vb, mask, attn);
  gemm_ffn1<<<dim3(32, 32), 256, 0, stream>>>(attn, w1b, b1, hb);
  gemm_ffn2_ln<<<dim3(32, 1), 256, 0, stream>>>(hb, w2b, b2, gamma, beta, x, out);
}
